// MaskedMultiHeadAttention_8753143349333
// MI455X (gfx1250) — matrix operand order repaired and checked by reading, not compiled, not run on hardware
//
#include <hip/hip_runtime.h>
#include <hip/hip_bf16.h>
#include <math.h>

// ---------------- problem constants ----------------
#define BATCH   2
#define SEQ     4096
#define DM      512
#define NHEADS  8
#define HDIM    64
#define WIN     256
#define MTOT    (BATCH * SEQ)          // 8192 rows

typedef __attribute__((ext_vector_type(16))) _Float16 v16h;
typedef __attribute__((ext_vector_type(8)))  float    v8f;

// ---------------- WMMA wrapper ----------------
__device__ __forceinline__ v8f wmma16(v16h a, v16h b, v8f c) {
  // D = A(16x32 f16) * B(32x16 f16) + C(16x16 f32)
  return __builtin_amdgcn_wmma_f32_16x16x32_f16(
      /*neg_a=*/false, a, /*neg_b=*/false, b,
      /*c_mod=*/(short)0, c, /*reuse_a=*/false, /*reuse_b=*/false);
}

// ---------------- fragment loaders (contiguous k = 16*hi + s order, same in A and B) ----------------
// A 16x32 f16, row-major source with leading dim lda.
__device__ __forceinline__ v16h load_a16(const _Float16* A, int lda, int lane) {
  int r = lane & 15, hi = lane >> 4;
  const _Float16* base = A + r * lda;
  v16h a;
#pragma unroll
  for (int v = 0; v < 8; ++v) {
    int k = hi * 16 + v * 2;
    a[2 * v]     = base[k];
    a[2 * v + 1] = base[k + 1];
  }
  return a;
}

// Same A fragment but converting from fp32 source on the fly.
__device__ __forceinline__ v16h load_a_f32(const float* A, int lda, int lane) {
  int r = lane & 15, hi = lane >> 4;
  const float* base = A + r * lda;
  v16h a;
#pragma unroll
  for (int v = 0; v < 8; ++v) {
    int k = hi * 16 + v * 2;
    a[2 * v]     = (_Float16)base[k];
    a[2 * v + 1] = (_Float16)base[k + 1];
  }
  return a;
}

// B 32x16 f16 where B = W^T, W is [N,K] row-major (ptr at row n0, K offset kb).
__device__ __forceinline__ v16h load_bT(const _Float16* W, int ldw, int lane) {
  int n = lane & 15, hi = lane >> 4;
  const _Float16* base = W + n * ldw + hi * 16;
  v16h b;
#pragma unroll
  for (int v = 0; v < 8; ++v) {
    b[2 * v]     = base[2 * v];
    b[2 * v + 1] = base[2 * v + 1];
  }
  return b;
}

// B 32x16 f16 from a row-major [K,N] source (V matrix), ptr at (k0, n0).
__device__ __forceinline__ v16h load_b_rm(const _Float16* Bm, int ldb, int lane) {
  int n = lane & 15, hi = lane >> 4;
  const _Float16* base = Bm + (hi * 16) * ldb + n;
  v16h b;
#pragma unroll
  for (int v = 0; v < 8; ++v) {
    b[2 * v]     = base[(2 * v) * ldb];
    b[2 * v + 1] = base[(2 * v + 1) * ldb];
  }
  return b;
}

// ---------------- kernel 0: fp32 -> f16 convert ----------------
__global__ void f32_to_f16_kernel(const float* __restrict__ in,
                                  _Float16* __restrict__ out, int n) {
  int i = blockIdx.x * blockDim.x + threadIdx.x;
  if (i < n) out[i] = (_Float16)in[i];
}

// ---------------- kernel 1: QKV projection ----------------
// Each wave computes a 16x64 strip of C = x @ qkv_w^T + b (register blocked:
// one A-frag feeds 4 WMMAs per K-step). The 64-wide strip is head-aligned.
__global__ __launch_bounds__(256) void qkv_gemm_kernel(
    const float* __restrict__ x, const _Float16* __restrict__ wh,
    const float* __restrict__ bias, _Float16* __restrict__ qh,
    _Float16* __restrict__ kh, _Float16* __restrict__ vh) {
  int wave = threadIdx.x >> 5, lane = threadIdx.x & 31;
  int tile = blockIdx.x * 8 + wave;
  const int TN = (3 * DM) / 64;  // 24 strip columns
  int tm = tile / TN, tn = tile % TN;

  const float* A = x + (size_t)tm * 16 * DM;
  const _Float16* W = wh + (size_t)tn * 64 * DM;

  v8f c0 = {}, c1 = {}, c2 = {}, c3 = {};
  for (int kb = 0; kb < DM; kb += 32) {
    v16h a = load_a_f32(A + kb, DM, lane);
    c0 = wmma16(a, load_bT(W + 0 * 16 * DM + kb, DM, lane), c0);
    c1 = wmma16(a, load_bT(W + 1 * 16 * DM + kb, DM, lane), c1);
    c2 = wmma16(a, load_bT(W + 2 * 16 * DM + kb, DM, lane), c2);
    c3 = wmma16(a, load_bT(W + 3 * 16 * DM + kb, DM, lane), c3);
  }
  v8f cs[4] = {c0, c1, c2, c3};

  int col = lane & 15, hi = lane >> 4;
  int n0 = tn * 64;                 // 64-aligned -> exactly one head, one section
  int sec = n0 >> 9;                // 0=q 1=k 2=v
  int h = (n0 & 511) >> 6;
  _Float16* dst = (sec == 0) ? qh : (sec == 1) ? kh : vh;
  float scl = (sec == 0) ? 0.125f : 1.0f;  // fold 1/sqrt(64) into Q

#pragma unroll
  for (int t = 0; t < 4; ++t) {
    float bv = bias[n0 + 16 * t + col];
    int d = 16 * t + col;
#pragma unroll
    for (int i = 0; i < 8; ++i) {
      int mrow = tm * 16 + 8 * hi + i;
      int b_ = mrow >> 12, s_ = mrow & (SEQ - 1);
      dst[(((size_t)(b_ * NHEADS + h) * SEQ) + s_) * HDIM + d] =
          (_Float16)((cs[t][i] + bv) * scl);
    }
  }
}

// ---------------- kernel 2: sliding-window flash attention ----------------
// One wave = one 16-query tile of one (b,h). Keys in chunks of 64:
// 8 score WMMAs + 8 PV WMMAs per chunk, softmax amortized over 64 keys.
__global__ __launch_bounds__(256) void attn_kernel(
    const _Float16* __restrict__ qb, const _Float16* __restrict__ kb_,
    const _Float16* __restrict__ vb, _Float16* __restrict__ attn) {
  __shared__ _Float16 plds[8][16 * 64];  // per-wave P staging (16 KB total)
  int wave = threadIdx.x >> 5, lane = threadIdx.x & 31;
  int tile = blockIdx.x * 8 + wave;      // 0..4095
  int t = tile & (SEQ / 16 - 1);
  int bh = tile >> 8;                    // b*8 + h
  int t0 = t * 16;

  const _Float16* Q = qb + (size_t)bh * SEQ * HDIM;
  const _Float16* K = kb_ + (size_t)bh * SEQ * HDIM;
  const _Float16* V = vb + (size_t)bh * SEQ * HDIM;

  int col = lane & 15, hi = lane >> 4;

  // Q fragments (head-dims 0..31 / 32..63) live in registers for the whole loop.
  v16h qa0 = load_a16(Q + (size_t)t0 * HDIM + 0, HDIM, lane);
  v16h qa1 = load_a16(Q + (size_t)t0 * HDIM + 32, HDIM, lane);

  v8f o[4] = {};
  float m[8], l[8];
#pragma unroll
  for (int i = 0; i < 8; ++i) { m[i] = -__builtin_inff(); l[i] = 0.0f; }

  _Float16* P = plds[wave];
  int jstart = (t0 >= WIN) ? ((t0 - WIN) & ~63) : 0;

  for (int j0 = jstart; j0 <= t0 + 15; j0 += 64) {
    // ---- scores: S[16 x 64] = Q @ K^T  (4 N-tiles x 2 K-steps = 8 WMMAs)
    v8f st[4];
#pragma unroll
    for (int n = 0; n < 4; ++n) {
      const _Float16* Kn = K + (size_t)(j0 + 16 * n) * HDIM;
      v8f s = {};
      s = wmma16(qa0, load_bT(Kn, HDIM, lane), s);
      s = wmma16(qa1, load_bT(Kn + 32, HDIM, lane), s);
      st[n] = s;
    }

    // ---- online softmax per query row (C layout: row = t0+8*hi+i, col lane&15)
#pragma unroll
    for (int i = 0; i < 8; ++i) {
      int qi = t0 + 8 * hi + i;
      float sv[4];
      float cm = -__builtin_inff();
#pragma unroll
      for (int n = 0; n < 4; ++n) {
        int j = j0 + 16 * n + col;
        bool valid = (j <= qi) && (j + WIN >= qi);
        sv[n] = valid ? st[n][i] : -__builtin_inff();
        cm = fmaxf(cm, sv[n]);
      }
#pragma unroll
      for (int off = 1; off < 16; off <<= 1)
        cm = fmaxf(cm, __shfl_xor(cm, off, 32));

      float mn = fmaxf(m[i], cm);
      float alpha, ps = 0.0f;
      if (mn == -__builtin_inff()) {
        alpha = 1.0f;
#pragma unroll
        for (int n = 0; n < 4; ++n) sv[n] = 0.0f;
      } else {
        alpha = __expf(m[i] - mn);  // m=-inf -> 0
#pragma unroll
        for (int n = 0; n < 4; ++n) {
          sv[n] = __expf(sv[n] - mn);  // masked (-inf) -> 0
          ps += sv[n];
        }
      }
#pragma unroll
      for (int off = 1; off < 16; off <<= 1)
        ps += __shfl_xor(ps, off, 32);

      l[i] = l[i] * alpha + ps;
      m[i] = mn;
#pragma unroll
      for (int n = 0; n < 4; ++n) o[n][i] *= alpha;

      // stage P (C layout -> row-major 16x64 in LDS for A-frag reload)
      int row = 8 * hi + i;
#pragma unroll
      for (int n = 0; n < 4; ++n)
        P[row * 64 + 16 * n + col] = (_Float16)sv[n];
    }

    // DS ops are in-order per wave; the clobber stops the compiler from
    // hoisting the A-frag loads above the cross-lane stores.
    asm volatile("s_wait_dscnt 0" ::: "memory");

    // ---- O += P(16x64) @ V(64x64): 2 K-halves x 4 d-tiles = 8 WMMAs
    v16h p0 = load_a16(P, 64, lane);        // P cols 0..31
    v16h p1 = load_a16(P + 32, 64, lane);   // P cols 32..63
    const _Float16* V0 = V + (size_t)j0 * HDIM;
    const _Float16* V1 = V + (size_t)(j0 + 32) * HDIM;
#pragma unroll
    for (int n = 0; n < 4; ++n) {
      o[n] = wmma16(p0, load_b_rm(V0 + 16 * n, HDIM, lane), o[n]);
      o[n] = wmma16(p1, load_b_rm(V1 + 16 * n, HDIM, lane), o[n]);
    }
  }

  // ---- normalize and store attn output as f16 [B][S][H*64]
  int b_ = bh >> 3, h_ = bh & 7;
#pragma unroll
  for (int i = 0; i < 8; ++i) {
    float inv = (l[i] > 0.0f) ? (1.0f / l[i]) : 0.0f;
    int s_ = t0 + 8 * hi + i;
    size_t base = ((size_t)b_ * SEQ + s_) * DM + h_ * HDIM + col;
#pragma unroll
    for (int n = 0; n < 4; ++n)
      attn[base + 16 * n] = (_Float16)(o[n][i] * inv);
  }
}

// ---------------- kernel 3: output projection ----------------
// Each wave computes a 16x64 strip of out = attn @ out_w^T + out_b (fp32 out).
__global__ __launch_bounds__(256) void out_gemm_kernel(
    const _Float16* __restrict__ attn, const _Float16* __restrict__ wh,
    const float* __restrict__ bias, float* __restrict__ out) {
  int wave = threadIdx.x >> 5, lane = threadIdx.x & 31;
  int tile = blockIdx.x * 8 + wave;
  const int TN = DM / 64;  // 8 strip columns
  int tm = tile / TN, tn = tile % TN;

  const _Float16* A = attn + (size_t)tm * 16 * DM;
  const _Float16* W = wh + (size_t)tn * 64 * DM;

  v8f c0 = {}, c1 = {}, c2 = {}, c3 = {};
  for (int kb = 0; kb < DM; kb += 32) {
    v16h a = load_a16(A + kb, DM, lane);
    c0 = wmma16(a, load_bT(W + 0 * 16 * DM + kb, DM, lane), c0);
    c1 = wmma16(a, load_bT(W + 1 * 16 * DM + kb, DM, lane), c1);
    c2 = wmma16(a, load_bT(W + 2 * 16 * DM + kb, DM, lane), c2);
    c3 = wmma16(a, load_bT(W + 3 * 16 * DM + kb, DM, lane), c3);
  }
  v8f cs[4] = {c0, c1, c2, c3};

  int col = lane & 15, hi = lane >> 4;
  int n0 = tn * 64;
#pragma unroll
  for (int tt = 0; tt < 4; ++tt) {
    int n = n0 + 16 * tt + col;
    float bv = bias[n];
#pragma unroll
    for (int i = 0; i < 8; ++i) {
      int mrow = tm * 16 + 8 * hi + i;
      out[(size_t)mrow * DM + n] = cs[tt][i] + bv;
    }
  }
}

// ---------------- launch ----------------
extern "C" void kernel_launch(void* const* d_in, const int* in_sizes, int n_in,
                              void* d_out, int out_size, void* d_ws, size_t ws_size,
                              hipStream_t stream) {
  (void)in_sizes; (void)n_in; (void)out_size; (void)ws_size;
  const float* x     = (const float*)d_in[0];
  const float* qkv_w = (const float*)d_in[1];
  const float* qkv_b = (const float*)d_in[2];
  const float* out_w = (const float*)d_in[3];
  const float* out_b = (const float*)d_in[4];
  float* out = (float*)d_out;

  // workspace layout (all f16): weights 2 MB, then q/k/v/attn 8 MB each = 34 MB
  char* ws = (char*)d_ws;
  _Float16* qkvw_h = (_Float16*)ws;                                  // 1536*512
  _Float16* outw_h = (_Float16*)(ws + (size_t)1536 * 512 * 2);       // 512*512
  _Float16* qh   = (_Float16*)(ws + (size_t)2 * 1024 * 1024);
  _Float16* kh   = qh + (size_t)BATCH * NHEADS * SEQ * HDIM;
  _Float16* vh   = kh + (size_t)BATCH * NHEADS * SEQ * HDIM;
  _Float16* attn = vh + (size_t)BATCH * NHEADS * SEQ * HDIM;

  f32_to_f16_kernel<<<(1536 * 512 + 255) / 256, 256, 0, stream>>>(qkv_w, qkvw_h, 1536 * 512);
  f32_to_f16_kernel<<<(512 * 512 + 255) / 256, 256, 0, stream>>>(out_w, outw_h, 512 * 512);

  // 512 row-tiles x 24 strip-columns = 12288 wave strips, 8 waves/block
  qkv_gemm_kernel<<<(MTOT / 16) * ((3 * DM) / 64) / 8, 256, 0, stream>>>(
      x, qkvw_h, qkv_b, qh, kh, vh);

  // B*H*(S/16) = 4096 wave tiles
  attn_kernel<<<(BATCH * NHEADS * (SEQ / 16)) / 8, 256, 0, stream>>>(qh, kh, vh, attn);

  // 512 x 8 = 4096 wave strips
  out_gemm_kernel<<<(MTOT / 16) * (DM / 64) / 8, 256, 0, stream>>>(attn, outw_h, out_b, out);
}
